// NetVLAD_55886114455954
// MI455X (gfx1250) — matrix operand order repaired and checked by reading, not compiled, not run on hardware
//
#include <hip/hip_runtime.h>

#define B_ 64
#define M_ 512
#define D_ 2048
#define K_ 32

typedef __attribute__((ext_vector_type(16))) __bf16          v16bf;
typedef __attribute__((ext_vector_type(8)))  float           v8f;
typedef __attribute__((ext_vector_type(8)))  unsigned short  u16x8;
typedef __attribute__((ext_vector_type(16))) unsigned short  u16x16;
typedef __attribute__((ext_vector_type(4)))  int             v4i;

union BF16Frag {
  u16x16 u;
  u16x8  h[2];
  unsigned short e[16];
  v16bf  b;
};

__device__ __forceinline__ unsigned short f2bf(float f) {
  unsigned int u = __float_as_uint(f);
  u += 0x7FFFu + ((u >> 16) & 1u);          // round-to-nearest-even
  return (unsigned short)(u >> 16);
}
__device__ __forceinline__ float bf2f(unsigned short s) {
  return __uint_as_float(((unsigned int)s) << 16);
}

// ---------------------------------------------------------------------------
// XOR-butterfly reductions via ds_swizzle_b32 (group-of-32: imm = xor<<10 | 0x1f)
// ---------------------------------------------------------------------------
template <int XM>
__device__ __forceinline__ float swz_xor(float v) {
  return __int_as_float(
      __builtin_amdgcn_ds_swizzle(__float_as_int(v), (XM << 10) | 0x1F));
}
__device__ __forceinline__ float wave_red_sum(float v) {   // 32 lanes
  v += swz_xor<16>(v); v += swz_xor<8>(v); v += swz_xor<4>(v);
  v += swz_xor<2>(v);  v += swz_xor<1>(v);
  return v;
}
__device__ __forceinline__ float half_red_sum(float v) {   // 16-lane groups
  v += swz_xor<8>(v); v += swz_xor<4>(v); v += swz_xor<2>(v); v += swz_xor<1>(v);
  return v;
}
__device__ __forceinline__ float half_red_max(float v) {
  v = fmaxf(v, swz_xor<8>(v)); v = fmaxf(v, swz_xor<4>(v));
  v = fmaxf(v, swz_xor<2>(v)); v = fmaxf(v, swz_xor<1>(v));
  return v;
}

// ---------------------------------------------------------------------------
// Async global->LDS staging (gfx1250 GLOBAL_LOAD_ASYNC_TO_LDS_B128, ASYNCcnt)
// ---------------------------------------------------------------------------
#if defined(__has_builtin)
#if __has_builtin(__builtin_amdgcn_global_load_async_to_lds_b128)
#define HAVE_ASYNC_LDS 1
#endif
#if __has_builtin(__builtin_amdgcn_s_wait_asynccnt)
#define HAVE_WAIT_ASYNC_BUILTIN 1
#endif
#endif

#if defined(HAVE_ASYNC_LDS)
typedef __attribute__((address_space(1))) v4i* gl_v4i_ptr;   // global (__device__)
typedef __attribute__((address_space(3))) v4i* ld_v4i_ptr;   // LDS
#endif

__device__ __forceinline__ void wait_asynccnt0() {
#if defined(HAVE_WAIT_ASYNC_BUILTIN)
  __builtin_amdgcn_s_wait_asynccnt(0);
#else
  asm volatile("s_wait_asynccnt 0x0" ::: "memory");
#endif
}

// ---------------------------------------------------------------------------
// Kernel 1: per-frame L2 normalize x [B*M, D] f32 -> xn bf16
// ---------------------------------------------------------------------------
__global__ __launch_bounds__(256) void k_normalize(const float* __restrict__ x,
                                                   unsigned short* __restrict__ xnbf) {
  __shared__ float sh[8];
  const size_t row = blockIdx.x;                 // b*M + m
  const float* xr = x + row * (size_t)D_;
  unsigned short* orow = xnbf + row * (size_t)D_;
  float v[8];
  float ss = 0.f;
#pragma unroll
  for (int i = 0; i < 8; ++i) {
    v[i] = xr[threadIdx.x + i * 256];
    ss += v[i] * v[i];
  }
  ss = wave_red_sum(ss);
  if ((threadIdx.x & 31) == 0) sh[threadIdx.x >> 5] = ss;
  __syncthreads();
  if (threadIdx.x == 0) {
    float t = 0.f;
#pragma unroll
    for (int i = 0; i < 8; ++i) t += sh[i];
    sh[0] = 1.0f / fmaxf(sqrtf(t), 1e-12f);
  }
  __syncthreads();
  const float rn = sh[0];
#pragma unroll
  for (int i = 0; i < 8; ++i)
    orow[threadIdx.x + i * 256] = f2bf(v[i] * rn);
}

// ---------------------------------------------------------------------------
// Kernel 2: W f32 -> bf16 ; zero batch-norm accumulators
// ---------------------------------------------------------------------------
__global__ __launch_bounds__(256) void k_convW(const float* __restrict__ W,
                                               unsigned short* __restrict__ Wbf,
                                               float* __restrict__ bacc) {
  const int i = blockIdx.x * 1024 + threadIdx.x;   // 64 blocks cover K*D = 65536
#pragma unroll
  for (int j = 0; j < 4; ++j)
    Wbf[i + j * 256] = f2bf(W[i + j * 256]);
  if (blockIdx.x == 0 && threadIdx.x < B_) bacc[threadIdx.x] = 0.f;
}

// ---------------------------------------------------------------------------
// Kernel 3: logits = xn @ W^T via WMMA, softmax over K=32 in-register,
//           store assignments transposed saT[B,K,M] bf16.
// One wave handles a 16(m) x 32(k) tile.
// ---------------------------------------------------------------------------
__global__ __launch_bounds__(256) void k_logits_softmax(
    const unsigned short* __restrict__ xnbf,
    const unsigned short* __restrict__ Wbf,
    unsigned short* __restrict__ saT) {
  const int wave = threadIdx.x >> 5;
  const int lane = threadIdx.x & 31;
  const int half = lane >> 4;
  const int l15  = lane & 15;
  const int tile = blockIdx.x * 8 + wave;        // 2048 tiles total
  const int b    = tile >> 5;                    // 32 tiles per batch
  const int m0   = (tile & 31) << 4;

  const unsigned short* xr = xnbf + ((size_t)(b * M_ + m0 + l15)) * D_;
  const unsigned short* w0 = Wbf + (size_t)l15 * D_ + half * 16;          // k = l15
  const unsigned short* w1 = Wbf + (size_t)(16 + l15) * D_ + half * 16;   // k = 16+l15

  v8f c0 = {};
  v8f c1 = {};
#pragma unroll 2
  for (int d0 = 0; d0 < D_; d0 += 32) {
    __builtin_prefetch(xr + d0 + 512, 0, 3);     // global_prefetch ~1KB ahead
    BF16Frag a, b0, b1;
    // A 16x32 bf16: lane-half selects K-subset 0..15 / 16..31, contiguous in d
    a.h[0] = *(const u16x8*)(xr + d0 + half * 16);
    a.h[1] = *(const u16x8*)(xr + d0 + half * 16 + 8);
    // B 32x16 bf16: lane-half selects red 0..15 / 16..31, contiguous in d
    b0.h[0] = *(const u16x8*)(w0 + d0);
    b0.h[1] = *(const u16x8*)(w0 + d0 + 8);
    b1.h[0] = *(const u16x8*)(w1 + d0);
    b1.h[1] = *(const u16x8*)(w1 + d0 + 8);
    c0 = __builtin_amdgcn_wmma_f32_16x16x32_bf16(false, a.b, false, b0.b,
                                                 (short)0, c0, false, false);
    c1 = __builtin_amdgcn_wmma_f32_16x16x32_bf16(false, a.b, false, b1.b,
                                                 (short)0, c1, false, false);
  }

  // C layout: VGPR v holds row (v + half*8), col = l15 (c0: k 0..15, c1: k 16..31)
  unsigned short* sp = saT + (size_t)b * K_ * M_;
#pragma unroll
  for (int j = 0; j < 8; ++j) {
    const float mx = half_red_max(fmaxf(c0[j], c1[j]));
    const float e0 = __expf(c0[j] - mx);
    const float e1 = __expf(c1[j] - mx);
    const float sm = half_red_sum(e0 + e1);
    const float inv = 1.0f / sm;                 // sm >= 1, safe
    const int m = m0 + j + half * 8;
    sp[(size_t)l15 * M_ + m]        = f2bf(e0 * inv);
    sp[(size_t)(16 + l15) * M_ + m] = f2bf(e1 * inv);
  }
}

// ---------------------------------------------------------------------------
// Kernel 4: colsum[b,k] = sum_m sa[b,m,k]   (rows of saT are contiguous)
// ---------------------------------------------------------------------------
__global__ __launch_bounds__(32) void k_colsum(const unsigned short* __restrict__ saT,
                                               float* __restrict__ colsum) {
  const size_t row = blockIdx.x;               // b*K + k
  const unsigned short* r = saT + row * M_;
  float s = 0.f;
#pragma unroll
  for (int i = 0; i < 16; ++i) s += bf2f(r[threadIdx.x + i * 32]);
  s = wave_red_sum(s);
  if (threadIdx.x == 0) colsum[row] = s;
}

// ---------------------------------------------------------------------------
// Kernel 5: agg[b,k,d] = sum_m sa[b,m,k]*xn[b,m,d] via WMMA.
// Block = (batch b, 256-wide d tile); xn tiles staged in LDS (async direct
// global->LDS path when available); A fragments contiguous from saT.
// ---------------------------------------------------------------------------
__global__ __launch_bounds__(256) void k_agg(
    const unsigned short* __restrict__ xnbf,
    const unsigned short* __restrict__ saT,
    float* __restrict__ out) {
  __shared__ unsigned short tile[32][264];     // 32 m x 256 d (+pad), 16.5 KB
  const int wave  = threadIdx.x >> 5;
  const int lane  = threadIdx.x & 31;
  const int half  = lane >> 4;
  const int l15   = lane & 15;
  const int b     = blockIdx.x >> 3;
  const int dbase = (blockIdx.x & 7) << 8;     // 8 d-tiles of 256
  const int dloc  = wave << 5;                 // this wave's 32 columns in LDS

  v8f c00 = {}, c01 = {}, c10 = {}, c11 = {};
  const unsigned short* sp = saT + (size_t)b * K_ * M_;

  for (int mc = 0; mc < M_; mc += 32) {
    // stage xn[mc..mc+31, dbase..dbase+255]
#pragma unroll
    for (int r = 0; r < 4; ++r) {
      const int row = (threadIdx.x >> 5) + r * 8;
      const int col = (threadIdx.x & 31) * 8;
      const unsigned short* gp =
          xnbf + ((size_t)(b * M_ + mc + row)) * D_ + dbase + col;
      __builtin_prefetch(gp + 32 * D_, 0, 3);  // next m-chunk
#if defined(HAVE_ASYNC_LDS)
      __builtin_amdgcn_global_load_async_to_lds_b128(
          (gl_v4i_ptr)(uintptr_t)gp,
          (ld_v4i_ptr)(unsigned)(uintptr_t)&tile[row][col],
          0, 0);
#else
      *(u16x8*)&tile[row][col] = *(const u16x8*)gp;
#endif
    }
#if defined(HAVE_ASYNC_LDS)
    wait_asynccnt0();
#endif
    __syncthreads();

    // A = sa^T [k x m]: rows k, reduction m (contiguous in saT)
    BF16Frag a0, a1;
    const int mo = mc + half * 16;
    a0.h[0] = *(const u16x8*)(sp + (size_t)l15 * M_ + mo);
    a0.h[1] = *(const u16x8*)(sp + (size_t)l15 * M_ + mo + 8);
    a1.h[0] = *(const u16x8*)(sp + (size_t)(16 + l15) * M_ + mo);
    a1.h[1] = *(const u16x8*)(sp + (size_t)(16 + l15) * M_ + mo + 8);

    // B = xn [m x d]: gather transposed pattern from LDS
    BF16Frag b0, b1;
#pragma unroll
    for (int j = 0; j < 16; ++j) {
      b0.e[j] = tile[half * 16 + j][dloc + l15];
      b1.e[j] = tile[half * 16 + j][dloc + 16 + l15];
    }

    c00 = __builtin_amdgcn_wmma_f32_16x16x32_bf16(false, a0.b, false, b0.b,
                                                  (short)0, c00, false, false);
    c01 = __builtin_amdgcn_wmma_f32_16x16x32_bf16(false, a0.b, false, b1.b,
                                                  (short)0, c01, false, false);
    c10 = __builtin_amdgcn_wmma_f32_16x16x32_bf16(false, a1.b, false, b0.b,
                                                  (short)0, c10, false, false);
    c11 = __builtin_amdgcn_wmma_f32_16x16x32_bf16(false, a1.b, false, b1.b,
                                                  (short)0, c11, false, false);
    __syncthreads();
  }

  const int dcol = dbase + dloc;
#pragma unroll
  for (int v = 0; v < 8; ++v) {
    const int k0 = v + half * 8;
    const size_t base0 = ((size_t)(b * K_ + k0)) * D_ + dcol;
    const size_t base1 = ((size_t)(b * K_ + 16 + k0)) * D_ + dcol;
    out[base0 + l15]      = c00[v];
    out[base0 + 16 + l15] = c01[v];
    out[base1 + l15]      = c10[v];
    out[base1 + 16 + l15] = c11[v];
  }
}

// ---------------------------------------------------------------------------
// Kernel 6a: vlad = agg - colsum*centroid; intra-normalize per (b,k);
//            accumulate per-batch sumsq of normalized rows.
// ---------------------------------------------------------------------------
__global__ __launch_bounds__(256) void k_intranorm(
    float* __restrict__ out, const float* __restrict__ cent,
    const float* __restrict__ colsum, float* __restrict__ bacc) {
  __shared__ float sh[8];
  const int b = blockIdx.x >> 5;
  const int k = blockIdx.x & 31;
  float* o = out + ((size_t)(b * K_ + k)) * D_;
  const float* c = cent + (size_t)k * D_;
  const float cs = colsum[b * K_ + k];
  float v[8];
  float ss = 0.f;
#pragma unroll
  for (int i = 0; i < 8; ++i) {
    const int d = threadIdx.x + i * 256;
    v[i] = o[d] - cs * c[d];
    ss += v[i] * v[i];
  }
  ss = wave_red_sum(ss);
  if ((threadIdx.x & 31) == 0) sh[threadIdx.x >> 5] = ss;
  __syncthreads();
  if (threadIdx.x == 0) {
    float t = 0.f;
#pragma unroll
    for (int i = 0; i < 8; ++i) t += sh[i];
    const float rn = 1.0f / fmaxf(sqrtf(t), 1e-12f);
    sh[0] = rn;
    atomicAdd(&bacc[b], t * rn * rn);
  }
  __syncthreads();
  const float rn = sh[0];
#pragma unroll
  for (int i = 0; i < 8; ++i)
    o[threadIdx.x + i * 256] = v[i] * rn;
}

// ---------------------------------------------------------------------------
// Kernel 6b: final per-batch L2 normalize over K*D
// ---------------------------------------------------------------------------
__global__ __launch_bounds__(256) void k_finalnorm(float* __restrict__ out,
                                                   const float* __restrict__ bacc) {
  const int b = blockIdx.x >> 5;
  const size_t base = (size_t)b * (K_ * D_) + (size_t)(blockIdx.x & 31) * 2048;
  const float s = 1.0f / fmaxf(sqrtf(bacc[b]), 1e-12f);
#pragma unroll
  for (int i = 0; i < 8; ++i)
    out[base + threadIdx.x + i * 256] *= s;
}

// ---------------------------------------------------------------------------
extern "C" void kernel_launch(void* const* d_in, const int* in_sizes, int n_in,
                              void* d_out, int out_size, void* d_ws, size_t ws_size,
                              hipStream_t stream) {
  const float* x    = (const float*)d_in[0];   // [B, M, D]
  const float* W    = (const float*)d_in[1];   // [K, D]
  const float* cent = (const float*)d_in[2];   // [K, D]
  float* out = (float*)d_out;                  // [B, K*D]

  // workspace layout
  unsigned short* xnbf = (unsigned short*)d_ws;          // B*M*D bf16 (128 MB)
  unsigned short* Wbf  = xnbf + (size_t)B_ * M_ * D_;    // K*D bf16
  unsigned short* saT  = Wbf + (size_t)K_ * D_;          // B*K*M bf16 (2 MB)
  float* colsum = (float*)(saT + (size_t)B_ * K_ * M_);  // B*K f32
  float* bacc   = colsum + B_ * K_;                      // B f32

  k_normalize<<<B_ * M_, 256, 0, stream>>>(x, xnbf);
  k_convW<<<64, 256, 0, stream>>>(W, Wbf, bacc);
  k_logits_softmax<<<(B_ * M_ / 16) / 8, 256, 0, stream>>>(xnbf, Wbf, saT);
  k_colsum<<<B_ * K_, 32, 0, stream>>>(saT, colsum);
  k_agg<<<B_ * 8, 256, 0, stream>>>(xnbf, saT, out);
  k_intranorm<<<B_ * K_, 256, 0, stream>>>(out, cent, colsum, bacc);
  k_finalnorm<<<B_ * 32, 256, 0, stream>>>(out, bacc);
}
